// LocalizedLoraLayer_84971632984467
// MI455X (gfx1250) — compile-verified, matrix operand order checked by reading, not run on hardware
//
#include <hip/hip_runtime.h>

// ---------------------------------------------------------------------------
// x[4,2048,4096] f32, W[4096,4096] f32, b[4096], A[8,8,16,512], B[8,8,512,16]
// out = x @ W^T + b + (alpha/r) * (x @ Weff^T)
// Weff[j*512+o][k*512+i] = sum_r B[k][j][o][r] * A[k][j][r][i]
// Pass 1a folds everything into Wtot (bf16, ws). Pass 1b converts x to bf16
// (ws). Pass 2 is one bf16-WMMA GEMM (M=8192,N=4096,K=4096) with BOTH tiles
// staged via global_load_async_to_lds_b128, double-buffered LDS.
// ---------------------------------------------------------------------------

#define D      4096
#define MROWS  8192
#define BM     128
#define BN     256
#define KT     64
#define ROWB   144             // padded LDS row stride (72 bf16) -> no systematic bank conflicts
#define XBYTES (BM * ROWB)     // 18432
#define WBYTES (BN * ROWB)     // 36864
#define TILEB  (XBYTES + WBYTES)
#define NIT    (D / KT)        // 64

typedef __attribute__((ext_vector_type(16))) __bf16 v16bf;
typedef __attribute__((ext_vector_type(8)))  float  v8f;
typedef __attribute__((ext_vector_type(4)))  int    v4i;

union Frag { v16bf v; uint4 u[2]; };

#if defined(__gfx1250__) && \
    __has_builtin(__builtin_amdgcn_global_load_async_to_lds_b128) && \
    __has_builtin(__builtin_amdgcn_s_wait_asynccnt)
#define USE_ASYNC 1
typedef __attribute__((address_space(1))) char  gchar_t;
typedef __attribute__((address_space(3))) char  lchar_t;
typedef __attribute__((address_space(1))) v4i*  gv4i_p;   // ptr to v4i in global AS
typedef __attribute__((address_space(3))) v4i*  lv4i_p;   // ptr to v4i in LDS AS
#else
#define USE_ASYNC 0
#endif

__device__ __forceinline__ unsigned short f2bf(float f) {
    union { float f; unsigned int u; } v; v.f = f;
    unsigned int u = v.u;
    u += 0x7fffu + ((u >> 16) & 1u);   // round-to-nearest-even
    return (unsigned short)(u >> 16);
}
// Packed f32x2 -> bf16x2 RNE: 2x (bfe+add3) + one v_perm_b32 merge.
__device__ __forceinline__ unsigned int pk2(float lo, float hi) {
    unsigned int ul = __float_as_uint(lo);
    unsigned int uh = __float_as_uint(hi);
    ul += 0x7fffu + ((ul >> 16) & 1u);
    uh += 0x7fffu + ((uh >> 16) & 1u);
    return __builtin_amdgcn_perm(uh, ul, 0x07060302);  // {uh[3],uh[2],ul[3],ul[2]}
}

// ---------------------------------------------------------------------------
// Pass 1a: Wtot = bf16(W + scale * Weff)
// ---------------------------------------------------------------------------
__global__ __launch_bounds__(256)
void build_wtot(const float* __restrict__ W,
                const float* __restrict__ A,
                const float* __restrict__ B,
                const int*   __restrict__ alpha,
                const int*   __restrict__ rblk,
                unsigned short* __restrict__ Wtot) {
    const unsigned int idx = blockIdx.x * 256u + threadIdx.x;
    const int og = idx >> 12, ig = idx & 4095;
    const int j = og >> 9, o = og & 511;
    const int k = ig >> 9, i = ig & 511;

    const float* Ar = A + ((size_t)((k * 8 + j) * 16) * 512 + i);
    const float* Br = B + ((size_t)((k * 8 + j) * 512 + o) * 16);

    float acc = 0.0f;
#pragma unroll
    for (int r = 0; r < 16; ++r)
        acc += Br[r] * Ar[(size_t)r * 512];

    const float scale = (float)alpha[0] / (float)rblk[0];
    Wtot[idx] = f2bf(W[idx] + scale * acc);
}

// ---------------------------------------------------------------------------
// Pass 1b: Xbf = bf16(x)   (one-time conversion; removes all conversion VALU
// from the GEMM inner loop)
// ---------------------------------------------------------------------------
__global__ __launch_bounds__(256)
void convert_x(const float* __restrict__ X, unsigned short* __restrict__ Xbf) {
    const size_t i4 = ((size_t)blockIdx.x * 256u + threadIdx.x) * 4;
    float4 f = *(const float4*)(X + i4);
    uint2 u; u.x = pk2(f.x, f.y); u.y = pk2(f.z, f.w);
    *(uint2*)(Xbf + i4) = u;
}

// ---------------------------------------------------------------------------
// Tile staging: rows of D bf16 (8 x 16B chunks per 64-element K-slice).
// chunk id = q*256 + tid; row = id>>3 (+32/q-step), col = id&7 -> every wave
// instruction covers contiguous global memory.
// ---------------------------------------------------------------------------
#if USE_ASYNC
template <int NQ>
__device__ __forceinline__ void stage_async(const unsigned short* src, int row0, int k0,
                                            unsigned char* buf, int tid) {
    gv4i_p gp = (gv4i_p)((gchar_t*)(unsigned short*)src +
        (size_t)(row0 + (tid >> 3)) * (D * 2) + (size_t)k0 * 2 + (tid & 7) * 16);
    lv4i_p lp = (lv4i_p)((lchar_t*)(buf + (tid >> 3) * ROWB + (tid & 7) * 16));
#pragma unroll
    for (int q = 0; q < NQ; ++q) {
        __builtin_amdgcn_global_load_async_to_lds_b128(gp, lp, 0, 0);
        gp += (32 * (D * 2)) / 16;   // +32 rows (v4i units)
        lp += (32 * ROWB) / 16;
    }
}
#else
template <int NQ>
__device__ __forceinline__ void stage_sync(const unsigned short* src, int row0, int k0,
                                           unsigned char* buf, int tid) {
    const uint4* gp = (const uint4*)((const char*)src +
        (size_t)(row0 + (tid >> 3)) * (D * 2) + (size_t)k0 * 2 + (tid & 7) * 16);
    uint4* lp = (uint4*)(buf + (tid >> 3) * ROWB + (tid & 7) * 16);
    uint4 r[NQ];
#pragma unroll
    for (int q = 0; q < NQ; ++q) { r[q] = *gp; gp += (32 * D * 2) / 16; }
#pragma unroll
    for (int q = 0; q < NQ; ++q) { *lp = r[q]; lp += (32 * ROWB) / 16; }
}
#endif

// fallback GEMM staging (convert-in-kernel path)
__device__ __forceinline__ void load_x(float4 xr[8], const float* X, int m0, int k0, int tid) {
    const float4* gp = (const float4*)((const char*)X +
        (size_t)(m0 + (tid >> 4)) * (D * 4) + (size_t)k0 * 4 + (tid & 15) * 16);
#pragma unroll
    for (int q = 0; q < 8; ++q) { xr[q] = *gp; gp += 16 * (D / 4); }
}
__device__ __forceinline__ void store_x(const float4 xr[8], unsigned char* bufX, int tid) {
    uint2* lp = (uint2*)(bufX + (tid >> 4) * ROWB + (tid & 15) * 8);
#pragma unroll
    for (int q = 0; q < 8; ++q) {
        uint2 u; u.x = pk2(xr[q].x, xr[q].y); u.y = pk2(xr[q].z, xr[q].w);
        *lp = u;
        lp += (16 * ROWB) / 8;
    }
}

// ---------------------------------------------------------------------------
// One K-tile of WMMA work (KT=64 -> 2 k-steps x 4x4 tiles)
// ---------------------------------------------------------------------------
__device__ __forceinline__ void compute_tile(v8f acc[4][4],
                                             const unsigned char* sx,
                                             const unsigned char* sw,
                                             int waveM, int waveN,
                                             int lrow, int lhalf) {
#pragma unroll
    for (int kk = 0; kk < KT; kk += 32) {
        // ISA 16-bit A/B fragment layout: lane<16 -> K{0..7,16..23},
        // lane>=16 -> K{8..15,24..31}: two 16B chunks 32B apart.
        const int co0 = (kk + lhalf * 8) * 2;
        Frag af[4], bf[4];
#pragma unroll
        for (int ms = 0; ms < 4; ++ms) {
            const unsigned char* p = sx + (waveM * 64 + ms * 16 + lrow) * ROWB + co0;
            af[ms].u[0] = *(const uint4*)p;
            af[ms].u[1] = *(const uint4*)(p + 32);
        }
#pragma unroll
        for (int ns = 0; ns < 4; ++ns) {
            const unsigned char* p = sw + (waveN * 64 + ns * 16 + lrow) * ROWB + co0;
            bf[ns].u[0] = *(const uint4*)p;
            bf[ns].u[1] = *(const uint4*)(p + 32);
        }
#pragma unroll
        for (int ms = 0; ms < 4; ++ms)
#pragma unroll
            for (int ns = 0; ns < 4; ++ns)
                acc[ms][ns] = __builtin_amdgcn_wmma_f32_16x16x32_bf16(
                    false, af[ms].v, false, bf[ns].v,
                    (short)0, acc[ms][ns], false, false);
    }
}

__device__ __forceinline__ void epilogue(const v8f acc[4][4], const float* bias, float* Out,
                                         int m0, int n0, int waveM, int waveN,
                                         int lrow, int lhalf) {
#pragma unroll
    for (int ms = 0; ms < 4; ++ms) {
        const int mbase = m0 + waveM * 64 + ms * 16 + lhalf * 8;
#pragma unroll
        for (int ns = 0; ns < 4; ++ns) {
            const int n = n0 + waveN * 64 + ns * 16 + lrow;
            const float bv = bias[n];
#pragma unroll
            for (int r = 0; r < 8; ++r)
                Out[(size_t)(mbase + r) * D + n] = acc[ms][ns][r] + bv;
        }
    }
}

// ---------------------------------------------------------------------------
// Pass 2 (fast): both operands pre-converted bf16; all staging async-to-LDS.
// ---------------------------------------------------------------------------
__global__ __launch_bounds__(256)
void lora_gemm_pre(const unsigned short* __restrict__ Xbf,
                   const unsigned short* __restrict__ Wtot,
                   const float* __restrict__ bias,
                   float* __restrict__ Out) {
    __shared__ __align__(16) unsigned char smem[2 * TILEB];   // 110592 B

    const int tid   = threadIdx.x;
    const int lane  = tid & 31;
    const int wave  = tid >> 5;
    const int waveM = wave >> 2;
    const int waveN = wave & 3;
    const int m0 = blockIdx.x * BM;
    const int n0 = blockIdx.y * BN;
    const int lrow  = lane & 15;
    const int lhalf = lane >> 4;

    v8f acc[4][4];
#pragma unroll
    for (int a = 0; a < 4; ++a)
#pragma unroll
        for (int b = 0; b < 4; ++b)
            acc[a][b] = (v8f)(0.0f);

    // prologue: tile 0 -> buffer 0
#if USE_ASYNC
    stage_async<4>(Xbf,  m0, 0, smem,          tid);
    stage_async<8>(Wtot, n0, 0, smem + XBYTES, tid);
    __builtin_amdgcn_s_wait_asynccnt(0);
#else
    stage_sync<4>(Xbf,  m0, 0, smem,          tid);
    stage_sync<8>(Wtot, n0, 0, smem + XBYTES, tid);
#endif
    __syncthreads();

    for (int it = 0; it < NIT; ++it) {
        unsigned char* cur = smem + (size_t)(it & 1) * TILEB;
        unsigned char* nxt = smem + (size_t)((it + 1) & 1) * TILEB;
        if (it + 1 < NIT) {
            const int k0n = (it + 1) * KT;
#if USE_ASYNC
            stage_async<4>(Xbf,  m0, k0n, nxt,          tid);
            stage_async<8>(Wtot, n0, k0n, nxt + XBYTES, tid);
#else
            stage_sync<4>(Xbf,  m0, k0n, nxt,          tid);
            stage_sync<8>(Wtot, n0, k0n, nxt + XBYTES, tid);
#endif
        }
        compute_tile(acc, cur, cur + XBYTES, waveM, waveN, lrow, lhalf);
#if USE_ASYNC
        __builtin_amdgcn_s_wait_asynccnt(0);
#endif
        __syncthreads();
    }

    epilogue(acc, bias, Out, m0, n0, waveM, waveN, lrow, lhalf);
}

// ---------------------------------------------------------------------------
// Pass 2 (fallback, small ws): convert X f32->bf16 while staging.
// ---------------------------------------------------------------------------
__global__ __launch_bounds__(256)
void lora_gemm_cvt(const float* __restrict__ X,
                   const unsigned short* __restrict__ Wtot,
                   const float* __restrict__ bias,
                   float* __restrict__ Out) {
    __shared__ __align__(16) unsigned char smem[2 * TILEB];

    const int tid   = threadIdx.x;
    const int lane  = tid & 31;
    const int wave  = tid >> 5;
    const int waveM = wave >> 2;
    const int waveN = wave & 3;
    const int m0 = blockIdx.x * BM;
    const int n0 = blockIdx.y * BN;
    const int lrow  = lane & 15;
    const int lhalf = lane >> 4;

    v8f acc[4][4];
#pragma unroll
    for (int a = 0; a < 4; ++a)
#pragma unroll
        for (int b = 0; b < 4; ++b)
            acc[a][b] = (v8f)(0.0f);

    float4 xr[8];

    load_x(xr, X, m0, 0, tid);
#if USE_ASYNC
    stage_async<8>(Wtot, n0, 0, smem + XBYTES, tid);
#else
    stage_sync<8>(Wtot, n0, 0, smem + XBYTES, tid);
#endif
    store_x(xr, smem, tid);
#if USE_ASYNC
    __builtin_amdgcn_s_wait_asynccnt(0);
#endif
    __syncthreads();

    for (int it = 0; it < NIT; ++it) {
        unsigned char* cur = smem + (size_t)(it & 1) * TILEB;
        unsigned char* nxt = smem + (size_t)((it + 1) & 1) * TILEB;
        const bool hasNext = (it + 1) < NIT;

        if (hasNext) {
            const int k0n = (it + 1) * KT;
            load_x(xr, X, m0, k0n, tid);
#if USE_ASYNC
            stage_async<8>(Wtot, n0, k0n, nxt + XBYTES, tid);
#else
            stage_sync<8>(Wtot, n0, k0n, nxt + XBYTES, tid);
#endif
        }
        compute_tile(acc, cur, cur + XBYTES, waveM, waveN, lrow, lhalf);
        if (hasNext)
            store_x(xr, nxt, tid);
#if USE_ASYNC
        __builtin_amdgcn_s_wait_asynccnt(0);
#endif
        __syncthreads();
    }

    epilogue(acc, bias, Out, m0, n0, waveM, waveN, lrow, lhalf);
}

// ---------------------------------------------------------------------------
extern "C" void kernel_launch(void* const* d_in, const int* in_sizes, int n_in,
                              void* d_out, int out_size, void* d_ws, size_t ws_size,
                              hipStream_t stream) {
    const float* x     = (const float*)d_in[0];
    const float* W     = (const float*)d_in[1];
    const float* bias  = (const float*)d_in[2];
    const float* A     = (const float*)d_in[3];
    const float* B     = (const float*)d_in[4];
    const int*   alpha = (const int*)d_in[5];
    const int*   rblk  = (const int*)d_in[6];
    float* out = (float*)d_out;

    const size_t WTOT_BYTES = (size_t)D * D * 2;         // 32 MiB
    const size_t XBF_BYTES  = (size_t)MROWS * D * 2;     // 64 MiB

    unsigned short* Wtot = (unsigned short*)d_ws;
    build_wtot<<<(D * D) / 256, 256, 0, stream>>>(W, A, B, alpha, rblk, Wtot);

    dim3 grid(MROWS / BM, D / BN);   // 64 x 16

    if (ws_size >= WTOT_BYTES + XBF_BYTES) {
        unsigned short* Xbf = (unsigned short*)((char*)d_ws + WTOT_BYTES);
        convert_x<<<(MROWS * D / 4) / 256, 256, 0, stream>>>(x, Xbf);
        lora_gemm_pre<<<grid, 256, 0, stream>>>(Xbf, Wtot, bias, out);
    } else {
        lora_gemm_cvt<<<grid, 256, 0, stream>>>(x, Wtot, bias, out);
    }
}
